// SS2D_90898687852703
// MI455X (gfx1250) — compile-verified, matrix operand order checked by reading, not run on hardware
//
#include <hip/hip_runtime.h>
#include <hip/hip_bf16.h>

// ---------- types ----------
typedef __bf16 bf16_t;
typedef __attribute__((ext_vector_type(16))) __bf16 bf16x16;
typedef __attribute__((ext_vector_type(8)))  __bf16 bf16x8;
typedef __attribute__((ext_vector_type(8)))  float  f32x8;

__device__ __forceinline__ bf16_t f2bf(float f) {
  unsigned u = __builtin_bit_cast(unsigned, f);
  unsigned r = (u + 0x7FFFu + ((u >> 16) & 1u)) >> 16;   // round-to-nearest-even
  return __builtin_bit_cast(bf16_t, (unsigned short)r);
}
__device__ __forceinline__ float bf2f(bf16_t v) {
  unsigned u = ((unsigned)__builtin_bit_cast(unsigned short, v)) << 16;
  return __builtin_bit_cast(float, u);
}
__device__ __forceinline__ float silu_f(float x) { return x / (1.f + __expf(-x)); }

// ---------- weight transpose+convert: w(K,N) f32 -> wt(Nalloc,Kpad) bf16 ----------
__global__ void k_wtrans(const float* __restrict__ w, bf16_t* __restrict__ wt,
                         int K, int N, int Nalloc, int Kpad) {
  int idx = blockIdx.x * blockDim.x + threadIdx.x;
  int total = Nalloc * Kpad;
  if (idx >= total) return;
  int k = idx % Kpad;
  int n = idx / Kpad;
  float v = (n < N && k < K) ? w[(size_t)k * N + n] : 0.f;
  wt[idx] = f2bf(v);
}

// ---------- gather 4 directional sequences: x(B,C,H,W) -> seq(4B, L, C) bf16 ------
__global__ void k_gather(const float* __restrict__ x, bf16_t* __restrict__ seq,
                         int B, int C, int H, int W) {
  int L = H * W;
  long long total = (long long)4 * B * L * C;
  long long idx = (long long)blockIdx.x * blockDim.x + threadIdx.x;
  if (idx >= total) return;
  int c = (int)(idx % C);
  int l = (int)((idx / C) % L);
  int n = (int)(idx / ((long long)C * L));
  int dir = n / B, b = n % B;
  int src;
  if (dir == 0) { src = l; }
  else if (dir == 1) { int h = l / W, w = l % W; src = h * W + (W - 1 - w); }
  else if (dir == 2) { int w = l / H, h = l % H; src = h * W + w; }
  else               { int w = l / H, h = l % H; src = (H - 1 - h) * W + w; }
  seq[idx] = f2bf(x[((size_t)b * C + c) * L + src]);
}

// ---------- GEMM: C[M,N] f32 = A[M,K] bf16 @ Bt[N,K]^T bf16 -------------------
// K % 32 == 0.  Wave tile = 16(M) x 32(N), 4 waves/block -> 128 cols per block.
// Bt has Nalloc >= gridDim.x*128 zero-padded rows => no guards in the hot loop.
__global__ void k_gemm_bf16(const bf16_t* __restrict__ A, const bf16_t* __restrict__ Bt,
                            float* __restrict__ C, int M, int N, int K,
                            int lda, int ldb, int ldc, int Nalloc) {
  int lane  = threadIdx.x & 31;
  int wave  = threadIdx.x >> 5;
  int n0    = (blockIdx.x * 4 + wave) * 32;
  int m0    = blockIdx.y * 16;
  if (n0 >= Nalloc) return;            // wave-uniform
  int row   = lane & 15;               // A: M idx | B: N idx | C/D: N idx
  int half8 = (lane >> 4) << 3;        // K-half offset for A/B, +8 rows for C/D

  const bf16_t* ap  = A  + (size_t)(m0 + row)      * lda + half8;
  const bf16_t* bp0 = Bt + (size_t)(n0 + row)      * ldb + half8;
  const bf16_t* bp1 = Bt + (size_t)(n0 + 16 + row) * ldb + half8;

  f32x8 acc0 = {0.f,0.f,0.f,0.f,0.f,0.f,0.f,0.f};
  f32x8 acc1 = {0.f,0.f,0.f,0.f,0.f,0.f,0.f,0.f};

  for (int kt = 0; kt < K; kt += 32) {
    bf16x8 al  = *(const bf16x8*)(ap);
    bf16x8 ah  = *(const bf16x8*)(ap + 16);
    bf16x8 b0l = *(const bf16x8*)(bp0);
    bf16x8 b0h = *(const bf16x8*)(bp0 + 16);
    bf16x8 b1l = *(const bf16x8*)(bp1);
    bf16x8 b1h = *(const bf16x8*)(bp1 + 16);
    bf16x16 a  = __builtin_shufflevector(al,  ah,  0,1,2,3,4,5,6,7,8,9,10,11,12,13,14,15);
    bf16x16 b0 = __builtin_shufflevector(b0l, b0h, 0,1,2,3,4,5,6,7,8,9,10,11,12,13,14,15);
    bf16x16 b1 = __builtin_shufflevector(b1l, b1h, 0,1,2,3,4,5,6,7,8,9,10,11,12,13,14,15);
    acc0 = __builtin_amdgcn_wmma_f32_16x16x32_bf16(false, a, false, b0, (short)0, acc0, false, false);
    acc1 = __builtin_amdgcn_wmma_f32_16x16x32_bf16(false, a, false, b1, (short)0, acc1, false, false);
    ap += 32; bp0 += 32; bp1 += 32;
  }
#pragma unroll
  for (int r = 0; r < 8; ++r) {
    int m = m0 + r + half8;
    int n = n0 + row;
    if (m < M && n      < N) C[(size_t)m * ldc + n]      = acc0[r];
    if (m < M && n + 16 < N) C[(size_t)m * ldc + n + 16] = acc1[r];
  }
}

// ---------- depthwise conv1d (k=3) + optional bias + SiLU, f32 in -> bf16 out ----
__global__ void k_dwconv_silu(const float* __restrict__ in, int inStride,
                              const float* __restrict__ w, const float* __restrict__ bias,
                              bf16_t* __restrict__ out,
                              int NS, int L, int D, int padL) {
  long long total = (long long)NS * L * D;
  long long idx = (long long)blockIdx.x * blockDim.x + threadIdx.x;
  if (idx >= total) return;
  int d = (int)(idx % D);
  int l = (int)((idx / D) % L);
  int n = (int)(idx / ((long long)D * L));
  float acc = bias ? bias[d] : 0.f;
#pragma unroll
  for (int k = 0; k < 3; ++k) {
    int t = l - padL + k;
    if (t >= 0 && t < L)
      acc += in[((size_t)n * L + t) * inStride + d] * w[d * 3 + k];
  }
  out[((size_t)n * L + l) * D + d] = f2bf(silu_f(acc));
}

// ---------- dtA staging: dbc[:, :12] f32 -> (rows,32) bf16 zero-padded ----------
__global__ void k_dta(const float* __restrict__ dbc, bf16_t* __restrict__ dtA,
                      long long rows) {
  long long idx = (long long)blockIdx.x * blockDim.x + threadIdx.x;
  if (idx >= rows * 32) return;
  int k = (int)(idx % 32);
  size_t r = (size_t)(idx / 32);
  dtA[idx] = f2bf((k < 12) ? dbc[r * 44 + k] : 0.f);
}

// ---------- dt = softplus(dt_pre + bias), in-place f32 ----------
__global__ void k_dt_act(float* __restrict__ dt, const float* __restrict__ b,
                         long long total, int D) {
  long long idx = (long long)blockIdx.x * blockDim.x + threadIdx.x;
  if (idx >= total) return;
  float v = dt[idx] + b[(int)(idx % D)];
  dt[idx] = (v > 20.f) ? v : __logf(1.f + __expf(v));
}

// ---------- selective scan: one block/sequence, one thread/channel --------------
// Fuses  ys = (scan + xm*D) * silu(z);  outputs bf16 for the next WMMA GEMM.
__global__ void k_scan(const float* __restrict__ dt, const bf16_t* __restrict__ xm,
                       const float* __restrict__ dbc, const float* __restrict__ xz,
                       const float* __restrict__ A_log, const float* __restrict__ Dp,
                       bf16_t* __restrict__ ys, int L) {
  const int Dm = 192, S = 16, DBC = 44, R = 12;
  int n = blockIdx.x;
  int d = threadIdx.x;                    // 0..191
  float Arow[S], h[S];
#pragma unroll
  for (int s = 0; s < S; ++s) { Arow[s] = -__expf(A_log[d * S + s]); h[s] = 0.f; }
  float Dd = Dp[d];
  __shared__ float sB[S], sC[S];
  for (int t = 0; t < L; ++t) {
    size_t base = (size_t)n * L + t;
    __syncthreads();
    if (threadIdx.x < S)          sB[threadIdx.x]     = dbc[base * DBC + R + threadIdx.x];
    else if (threadIdx.x < 2 * S) sC[threadIdx.x - S] = dbc[base * DBC + R + S + (threadIdx.x - S)];
    __syncthreads();
    float dtv = dt[base * Dm + d];
    float xv  = bf2f(xm[base * Dm + d]);
    float dx  = dtv * xv;
    float y   = 0.f;
#pragma unroll
    for (int s = 0; s < S; ++s) {
      h[s] = __expf(dtv * Arow[s]) * h[s] + dx * sB[s];
      y += h[s] * sC[s];
    }
    float zv = xz[base * 2 * Dm + Dm + d];
    ys[base * Dm + d] = f2bf((y + xv * Dd) * silu_f(zv));
  }
}

// ---------- outer gate: y2 = mamba_out * silu(x_gate) -> bf16 ----------
__global__ void k_gate(const float* __restrict__ mo, const float* __restrict__ proj,
                       bf16_t* __restrict__ y2, long long total) {
  long long idx = (long long)blockIdx.x * blockDim.x + threadIdx.x;
  if (idx >= total) return;
  int d = (int)(idx % 192);
  size_t row = (size_t)(idx / 192);
  y2[idx] = f2bf(mo[idx] * silu_f(proj[row * 384 + 192 + d]));
}

// ---------- merge 4 directions -> merged(B, L, D) bf16 row-major ----------
__global__ void k_merge(const bf16_t* __restrict__ y2, bf16_t* __restrict__ merged,
                        int B, int H, int W, int D) {
  int L = H * W;
  long long total = (long long)B * L * D;
  long long idx = (long long)blockIdx.x * blockDim.x + threadIdx.x;
  if (idx >= total) return;
  int d = (int)(idx % D);
  int l = (int)((idx / D) % L);
  int b = (int)(idx / ((long long)D * L));
  int h = l / W, w = l % W;
  size_t s0 = ((size_t)(0 * B + b) * L + (h * W + w))           * D + d;
  size_t s1 = ((size_t)(1 * B + b) * L + (h * W + (W - 1 - w))) * D + d;
  size_t s2 = ((size_t)(2 * B + b) * L + (w * H + h))           * D + d;
  size_t s3 = ((size_t)(3 * B + b) * L + (w * H + (H - 1 - h))) * D + d;
  merged[idx] = f2bf(bf2f(y2[s0]) + bf2f(y2[s1]) + bf2f(y2[s2]) + bf2f(y2[s3]));
}

// ---------- layernorm over C + residual: out(B,C,H,W) f32 ----------
__global__ void k_ln_res(const float* __restrict__ op, const float* __restrict__ x,
                         const float* __restrict__ lnw, const float* __restrict__ lnb,
                         const float* __restrict__ rs, float* __restrict__ out,
                         int B, int C, int H, int W) {
  int L = H * W;
  int idx = blockIdx.x * blockDim.x + threadIdx.x;
  if (idx >= B * L) return;
  int b = idx / L, l = idx % L;
  const float* row = op + (size_t)idx * C;
  float mu = 0.f;
  for (int c = 0; c < C; ++c) mu += row[c];
  mu /= (float)C;
  float var = 0.f;
  for (int c = 0; c < C; ++c) { float t = row[c] - mu; var += t * t; }
  var /= (float)C;
  float inv = rsqrtf(var + 1e-5f);
  float scale = rs[0];
  for (int c = 0; c < C; ++c) {
    float o = (row[c] - mu) * inv * lnw[c] + lnb[c];
    size_t xi = ((size_t)b * C + c) * L + l;
    out[xi] = x[xi] + scale * o;
  }
}

// ---------- host ----------
static inline int cdiv(int a, int b) { return (a + b - 1) / b; }

extern "C" void kernel_launch(void* const* d_in, const int* in_sizes, int n_in,
                              void* d_out, int out_size, void* d_ws, size_t ws_size,
                              hipStream_t stream) {
  const float* x        = (const float*)d_in[0];
  const float* w_in     = (const float*)d_in[1];
  const float* w_conv   = (const float*)d_in[2];
  const float* m_in_w   = (const float*)d_in[3];
  const float* m_conv_w = (const float*)d_in[4];
  const float* m_conv_b = (const float*)d_in[5];
  const float* m_xproj  = (const float*)d_in[6];
  const float* m_dt_w   = (const float*)d_in[7];
  const float* m_dt_b   = (const float*)d_in[8];
  const float* m_A_log  = (const float*)d_in[9];
  const float* m_D      = (const float*)d_in[10];
  const float* m_out_w  = (const float*)d_in[11];
  const float* w_out    = (const float*)d_in[12];
  const float* ln_w     = (const float*)d_in[13];
  const float* ln_b     = (const float*)d_in[14];
  const float* res_s    = (const float*)d_in[15];

  const int B = 4, C = 96, H = 64, W = 64, L = H * W, Di = 192;
  const int NS = 4 * B;                      // 16 sequences
  const long long ML = (long long)NS * L;    // 65536 rows

  // ---- workspace carve (256B aligned regions) ----
  char* base = (char*)d_ws;
  auto carve = [&](size_t bytes) -> char* {
    char* p = base; base += (bytes + 255) & ~(size_t)255; return p;
  };
  float*  proj   = (float*) carve((size_t)ML * 384 * 4);   // f32, alive until gate
  float*  xz     = (float*) carve((size_t)ML * 384 * 4);   // f32, z half alive thru scan
  float*  dbc    = (float*) carve((size_t)ML * 44 * 4);    // f32 -> reused as outp f32
  float*  dt_mo  = (float*) carve((size_t)ML * 192 * 4);   // dt f32 -> reused as mamba_out
  bf16_t* seq    = (bf16_t*)carve((size_t)ML * 96 * 2);    // bf16 -> reused as merged
  bf16_t* u_ys   = (bf16_t*)carve((size_t)ML * 192 * 2);   // u bf16 -> reused as ys
  bf16_t* xm_y2  = (bf16_t*)carve((size_t)ML * 192 * 2);   // xm bf16 -> reused as y2
  bf16_t* dtA    = (bf16_t*)carve((size_t)ML * 32 * 2);    // K-padded dt GEMM A
  // transposed bf16 weights (N padded to block coverage of 128 cols)
  bf16_t* w_in_t   = (bf16_t*)carve((size_t)384 * 96 * 2);   // N=384, K=96
  bf16_t* m_in_t   = (bf16_t*)carve((size_t)384 * 192 * 2);  // N=384, K=192
  bf16_t* xproj_t  = (bf16_t*)carve((size_t)128 * 192 * 2);  // N=44 ->128, K=192
  bf16_t* dtw_t    = (bf16_t*)carve((size_t)256 * 32 * 2);   // N=192->256, K=12->32
  bf16_t* outw_t   = (bf16_t*)carve((size_t)256 * 192 * 2);  // N=192->256, K=192
  bf16_t* wout_t   = (bf16_t*)carve((size_t)128 * 192 * 2);  // N=96 ->128, K=192

  auto wtrans = [&](const float* w, bf16_t* wt, int K, int N, int Nalloc, int Kpad) {
    int tot = Nalloc * Kpad;
    k_wtrans<<<cdiv(tot, 256), 256, 0, stream>>>(w, wt, K, N, Nalloc, Kpad);
  };
  auto gemm = [&](const bf16_t* A, const bf16_t* Bt, float* Cm,
                  int M, int N, int K, int lda, int ldb, int ldc, int Nalloc) {
    dim3 grid(cdiv(Nalloc, 128), M / 16);
    k_gemm_bf16<<<grid, 128, 0, stream>>>(A, Bt, Cm, M, N, K, lda, ldb, ldc, Nalloc);
  };

  // 0) weights -> bf16 transposed
  wtrans(w_in,    w_in_t,  96, 384, 384, 96);
  wtrans(m_in_w,  m_in_t, 192, 384, 384, 192);
  wtrans(m_xproj, xproj_t,192,  44, 128, 192);
  wtrans(m_dt_w,  dtw_t,   12, 192, 256, 32);
  wtrans(m_out_w, outw_t, 192, 192, 256, 192);
  wtrans(w_out,   wout_t, 192,  96, 128, 192);

  // 1) gather 4 scan directions -> bf16
  { long long tot = ML * C;
    k_gather<<<(int)((tot + 255) / 256), 256, 0, stream>>>(x, seq, B, C, H, W); }
  // 2) proj = seq @ w_in   (65536 x 384, K=96)
  gemm(seq, w_in_t, proj, (int)ML, 384, 96, 96, 96, 384, 384);
  // 3) u = silu(dwconv_same(proj[:, :192])) -> bf16
  { long long tot = ML * Di;
    k_dwconv_silu<<<(int)((tot + 255) / 256), 256, 0, stream>>>(
        proj, 384, w_conv, nullptr, u_ys, NS, L, Di, 1); }
  // 4) xz = u @ m_in_w   (65536 x 384, K=192)
  gemm(u_ys, m_in_t, xz, (int)ML, 384, 192, 192, 192, 384, 384);
  // 5) xm = silu(causal dwconv(xz[:, :192]) + bias) -> bf16
  { long long tot = ML * Di;
    k_dwconv_silu<<<(int)((tot + 255) / 256), 256, 0, stream>>>(
        xz, 384, m_conv_w, m_conv_b, xm_y2, NS, L, Di, 2); }
  // 6) dbc = xm @ m_xproj_w   (65536 x 44, K=192)
  gemm(xm_y2, xproj_t, dbc, (int)ML, 44, 192, 192, 192, 44, 128);
  // 7) stage dbc[:, :12] -> (rows, 32) bf16, zero-padded K
  { long long tot = ML * 32;
    k_dta<<<(int)((tot + 255) / 256), 256, 0, stream>>>(dbc, dtA, ML); }
  // 8) dt_pre = dtA @ m_dt_w   (65536 x 192, K=32 padded)
  gemm(dtA, dtw_t, dt_mo, (int)ML, 192, 32, 32, 32, 192, 256);
  // 9) dt = softplus(dt_pre + m_dt_b)
  { long long tot = ML * Di;
    k_dt_act<<<(int)((tot + 255) / 256), 256, 0, stream>>>(dt_mo, m_dt_b, tot, Di); }
  // 10) selective scan (+ D skip + silu(z) gate) -> ys bf16 (overwrites u)
  k_scan<<<NS, Di, 0, stream>>>(dt_mo, xm_y2, dbc, xz, m_A_log, m_D, u_ys, L);
  // 11) mamba_out = ys @ m_out_w   (65536 x 192, K=192)  (overwrites dt f32)
  gemm(u_ys, outw_t, dt_mo, (int)ML, 192, 192, 192, 192, 192, 256);
  // 12) y2 = mamba_out * silu(x_gate) -> bf16 (overwrites xm)
  { long long tot = ML * Di;
    k_gate<<<(int)((tot + 255) / 256), 256, 0, stream>>>(dt_mo, proj, xm_y2, tot); }
  // 13) merge 4 directions -> merged(B, L, 192) bf16 (overwrites seq)
  { long long tot = (long long)B * L * Di;
    k_merge<<<(int)((tot + 255) / 256), 256, 0, stream>>>(xm_y2, seq, B, H, W, Di); }
  // 14) outp = merged @ w_out   (16384 x 96, K=192)  (overwrites dbc f32)
  gemm(seq, wout_t, dbc, B * L, 96, 192, 192, 192, 96, 128);
  // 15) layernorm + residual -> d_out (B, C, H, W)
  { int tot = B * L;
    k_ln_res<<<cdiv(tot, 128), 128, 0, stream>>>(dbc, x, ln_w, ln_b, res_s,
                                                 (float*)d_out, B, C, H, W); }
}
